// Cross_MultiHead_Attention_58686433133026
// MI455X (gfx1250) — matrix operand order repaired and checked by reading, not compiled, not run on hardware
//
#include <hip/hip_runtime.h>
#include <hip/hip_bf16.h>
#include <stdint.h>

// ---------------------------------------------------------------------------
// Cross multi-head attention for gfx1250 (MI455X), f16 WMMA + f32 accumulate.
// cvt(f32->f16) -> Q/K/V projection GEMMs (WMMA, 16x64 wave tile) ->
// flash attention (TDM-staged K/V tiles in LDS, WMMA QK^T + online softmax +
// WMMA PV) -> output projection GEMM + bias.
// ---------------------------------------------------------------------------

typedef __attribute__((ext_vector_type(16))) _Float16 v16h;
typedef __attribute__((ext_vector_type(8)))  _Float16 v8h;
typedef __attribute__((ext_vector_type(4)))  _Float16 v4h;
typedef __attribute__((ext_vector_type(8)))  float    v8f;
typedef __attribute__((ext_vector_type(4)))  unsigned v4u;
typedef __attribute__((ext_vector_type(8)))  unsigned v8u;

static constexpr int BATCH = 2;
static constexpr int QLEN  = 2048;
static constexpr int KVLEN = 4096;
static constexpr int EMB   = 1024;
static constexpr int HEADS = 16;
static constexpr int HDIM  = 64;

// workspace layout (in _Float16 elements)
static constexpr size_t XB_N  = (size_t)BATCH * QLEN  * EMB;
static constexpr size_t CB_N  = (size_t)BATCH * KVLEN * EMB;
static constexpr size_t W_N   = (size_t)EMB * EMB;
static constexpr size_t OFF_XB = 0;
static constexpr size_t OFF_CB = OFF_XB + XB_N;
static constexpr size_t OFF_WQ = OFF_CB + CB_N;
static constexpr size_t OFF_WK = OFF_WQ + W_N;
static constexpr size_t OFF_WV = OFF_WK + W_N;
static constexpr size_t OFF_WO = OFF_WV + W_N;
static constexpr size_t OFF_QB = OFF_WO + W_N;
static constexpr size_t OFF_KB = OFF_QB + XB_N;
static constexpr size_t OFF_VT = OFF_KB + CB_N;
static constexpr size_t OFF_AO = OFF_VT + CB_N;

// ---------------------------------------------------------------------------
// WMMA fragment helpers (16-bit, wave32; A and B share contiguous k = 16h + e order)
// ---------------------------------------------------------------------------

// B fragment (32x16 KxN): lane = N, element e -> K = e + 16*(lane>=16).
static __device__ inline v16h ldB(const _Float16* p) {
  v8h lo = *reinterpret_cast<const v8h*>(p);
  v8h hi = *reinterpret_cast<const v8h*>(p + 8);
  return __builtin_shufflevector(lo, hi, 0,1,2,3,4,5,6,7,8,9,10,11,12,13,14,15);
}

// A fragment (16x32 MxK): lane m = lane&15, h = lane>>4.
// element e -> K = e + 16h (matches ldB).
static __device__ inline v16h ldA(const _Float16* rowp, int h) {
  v8h lo = *reinterpret_cast<const v8h*>(rowp + 16 * h);
  v8h hi = *reinterpret_cast<const v8h*>(rowp + 16 * h + 8);
  return __builtin_shufflevector(lo, hi, 0,1,2,3,4,5,6,7,8,9,10,11,12,13,14,15);
}

static __device__ inline v8f wmma(v16h a, v16h b, v8f c) {
  return __builtin_amdgcn_wmma_f32_16x16x32_f16(false, a, false, b, (short)0, c,
                                                false, false);
}

static __device__ inline float halfmax(float x) {
#pragma unroll
  for (int m = 1; m < 16; m <<= 1) x = fmaxf(x, __shfl_xor(x, m, 32));
  return x;
}
static __device__ inline float halfsum(float x) {
#pragma unroll
  for (int m = 1; m < 16; m <<= 1) x += __shfl_xor(x, m, 32);
  return x;
}

// ---------------------------------------------------------------------------
// Tensor Data Mover: 2-D tile load Global -> LDS (D# per CDNA5 ISA).
// tile_w elements per row (contiguous), tile_h rows, row stride in elements.
// tensor dims == tile dims (tile is fully in-bounds, no clipping needed).
// Inline asm form is portable across both toolchains.
// ---------------------------------------------------------------------------
static __device__ inline void tdm_load_2d(void* lds_dst, const void* gsrc,
                                          unsigned tile_w, unsigned tile_h,
                                          unsigned long long row_stride_elems) {
  unsigned long long ga = (unsigned long long)(uintptr_t)gsrc;
  v4u g0;
  g0[0] = 1u;                                   // count=1 (valid), user mode
  g0[1] = (unsigned)(uintptr_t)lds_dst;         // LDS byte address (flat low 32b)
  g0[2] = (unsigned)ga;                         // global_addr[31:0]
  g0[3] = (unsigned)((ga >> 32) & 0x01FFFFFFu)  // global_addr[56:32]
          | (2u << 30);                         // type=2 ("image")
  v8u g1;
  g1[0] = 1u << 16;                             // data_size=1 -> 2 bytes/elem
  g1[1] = (tile_w & 0xFFFFu) << 16;             // tensor_dim0[15:0] @ bits63:48
  g1[2] = (tile_w >> 16) | ((tile_h & 0xFFFFu) << 16);  // td0[31:16] | td1[15:0]
  g1[3] = (tile_h >> 16) | ((tile_w & 0xFFFFu) << 16);  // td1[31:16] | tile_dim0
  g1[4] = (tile_h & 0xFFFFu);                   // tile_dim1 (tile_dim2=0)
  g1[5] = (unsigned)row_stride_elems;           // tensor_dim0_stride[31:0]
  g1[6] = (unsigned)((row_stride_elems >> 32) & 0xFFFFu);  // stride[47:32]
  g1[7] = 0u;
  asm volatile("tensor_load_to_lds %0, %1" :: "s"(g0), "s"(g1) : "memory");
}

// ---------------------------------------------------------------------------
// f32 -> f16 conversion (vectorized x4)
// ---------------------------------------------------------------------------
__global__ __launch_bounds__(256) void cvt_f32_f16(const float* __restrict__ in,
                                                   _Float16* __restrict__ out,
                                                   int n4) {
  int i = blockIdx.x * blockDim.x + threadIdx.x;
  if (i < n4) {
    float4 v = reinterpret_cast<const float4*>(in)[i];
    v4h o;
    o[0] = (_Float16)v.x; o[1] = (_Float16)v.y;
    o[2] = (_Float16)v.z; o[3] = (_Float16)v.w;
    reinterpret_cast<v4h*>(out)[i] = o;
  }
}

// ---------------------------------------------------------------------------
// GEMM: Y = A(f16,[M,1024]) @ W(f16,[N,1024])^T
// block = 256 threads (8 waves), tile 32(M) x 256(N); wave tile 16x64
// (4 accumulators sharing one A fragment -> 4 WMMA per 32-K step).
// MODE 0: f16 row-major   MODE 1: f16 per-head transposed (V)   MODE 2: f32+bias
// ---------------------------------------------------------------------------
template <int MODE>
__global__ __launch_bounds__(256) void gemm_wmma(
    const _Float16* __restrict__ A, const _Float16* __restrict__ W,
    _Float16* __restrict__ outH, float* __restrict__ outF,
    const float* __restrict__ bias, int M, int N) {
  constexpr int K = 1024;
  const int w    = threadIdx.x >> 5;
  const int lane = threadIdx.x & 31;
  const int h    = lane >> 4;
  const int c    = lane & 15;
  const int nblk = N / 256;
  const int m0   = (blockIdx.x / nblk) * 32;
  const int n0   = (blockIdx.x % nblk) * 256;
  const int wm   = (w & 1) * 16;
  const int wn   = (w >> 1) * 64;

  v8f acc[4] = {{}, {}, {}, {}};
  const _Float16* arow = A + (size_t)(m0 + wm + c) * K;
  const _Float16* wr[4];
#pragma unroll
  for (int t = 0; t < 4; t++)
    wr[t] = W + (size_t)(n0 + wn + 16 * t + c) * K + 16 * h;

#pragma unroll 2
  for (int k0 = 0; k0 < K; k0 += 32) {
    v16h a = ldA(arow + k0, h);
#pragma unroll
    for (int t = 0; t < 4; t++) acc[t] = wmma(a, ldB(wr[t] + k0), acc[t]);
  }

  if (MODE == 0) {
#pragma unroll
    for (int t = 0; t < 4; t++) {
      const int n = n0 + wn + 16 * t + c;
#pragma unroll
      for (int r = 0; r < 8; r++)
        outH[(size_t)(m0 + wm + r + 8 * h) * N + n] = (_Float16)acc[t][r];
    }
  } else if (MODE == 1) {
    // vT[((b*HEADS + hh)*HDIM + dd)][kv], kv contiguous across r
    const int mb = m0 + wm + 8 * h;
    const int bb = mb >> 12;            // / KVLEN
    const int kv = mb & (KVLEN - 1);
#pragma unroll
    for (int t = 0; t < 4; t++) {
      const int n = n0 + wn + 16 * t + c;
      const int hh = n >> 6, dd = n & 63;
      v8h ph;
#pragma unroll
      for (int r = 0; r < 8; r++) ph[r] = (_Float16)acc[t][r];
      *reinterpret_cast<v8h*>(outH +
          (size_t)((bb * HEADS + hh) * HDIM + dd) * KVLEN + kv) = ph;
    }
  } else {
#pragma unroll
    for (int t = 0; t < 4; t++) {
      const int n = n0 + wn + 16 * t + c;
      const float bv = bias[n];
#pragma unroll
      for (int r = 0; r < 8; r++)
        outF[(size_t)(m0 + wm + r + 8 * h) * N + n] = acc[t][r] + bv;
    }
  }
}

// ---------------------------------------------------------------------------
// Flash attention. grid = B * HEADS * (QLEN/128), block = 256 (8 waves).
// All 8 waves share one (batch, head, kv-window): wave 0 TDM-stages the
// 32x64 K tile and 64x32 V^T tile into double-buffered LDS; each wave owns
// 16 q rows. Per 32-KV step: 4 WMMA for S = QK^T, online softmax (half-wave
// shfl_xor butterflies), P via wave-private LDS transpose, 4 WMMA for PV.
// ---------------------------------------------------------------------------
__global__ __launch_bounds__(256) void attn_wmma(
    const _Float16* __restrict__ qb, const _Float16* __restrict__ kb,
    const _Float16* __restrict__ vT, _Float16* __restrict__ ao) {
  __shared__ __align__(16) _Float16 kT[2][32][64];   // [buf][kv_rel][d]
  __shared__ __align__(16) _Float16 vl[2][64][32];   // [buf][d][kv_rel]
  __shared__ __align__(16) _Float16 plds[8][16][32]; // wave-private P tiles

  const int w    = threadIdx.x >> 5;
  const int lane = threadIdx.x & 31;
  const int h    = lane >> 4;
  const int c    = lane & 15;

  const int qblk = blockIdx.x & 15;
  const int hd   = (blockIdx.x >> 4) & 15;
  const int b    = blockIdx.x >> 8;

  const int qrow0 = b * QLEN + qblk * 128 + w * 16;
  const int hcol  = hd * HDIM;

  // Q fragments for this wave (16 rows x 64 head-dim = two K32 A-frags)
  const _Float16* qrow = qb + (size_t)(qrow0 + c) * EMB + hcol;
  const v16h aq0 = ldA(qrow, h);
  const v16h aq1 = ldA(qrow + 32, h);

  float mI[8], lI[8];
  v8f o0 = {}, o1 = {}, o2 = {}, o3 = {};
#pragma unroll
  for (int r = 0; r < 8; r++) { mI[r] = -1e30f; lI[r] = 0.f; }

  const _Float16* kglob = kb + (size_t)(b * KVLEN) * EMB + hcol;       // +kv*EMB
  const _Float16* vglob = vT + (size_t)((b * HEADS + hd) * HDIM) * KVLEN;  // +kv

  if (w == 0) {
    tdm_load_2d(&kT[0][0][0], kglob, HDIM, 32, EMB);
    tdm_load_2d(&vl[0][0][0], vglob, 32, HDIM, KVLEN);
  }
  int cur = 0;

  for (int kv0 = 0; kv0 < KVLEN; kv0 += 32) {
    if (w == 0) __builtin_amdgcn_s_wait_tensorcnt(0);  // current bufs landed
    __syncthreads();  // publish bufs; also: everyone done reading cur^1
    if (w == 0 && kv0 + 32 < KVLEN) {
      tdm_load_2d(&kT[cur ^ 1][0][0], kglob + (size_t)(kv0 + 32) * EMB,
                  HDIM, 32, EMB);
      tdm_load_2d(&vl[cur ^ 1][0][0], vglob + (kv0 + 32), 32, HDIM, KVLEN);
    }

    // ---- S = Q @ K^T * (1/8) : two 16x16 tiles from LDS K tile ----
    const _Float16* kr0 = &kT[cur][c][0] + 16 * h;
    const _Float16* kr1 = &kT[cur][16 + c][0] + 16 * h;
    v8f s0 = {}, s1 = {};
    s0 = wmma(aq0, ldB(kr0), s0);
    s0 = wmma(aq1, ldB(kr0 + 32), s0);
    s1 = wmma(aq0, ldB(kr1), s1);
    s1 = wmma(aq1, ldB(kr1 + 32), s1);
    s0 = s0 * 0.125f;   // 1/sqrt(HDIM)
    s1 = s1 * 0.125f;

    // ---- online softmax over these 32 columns ----
#pragma unroll
    for (int r = 0; r < 8; r++) {
      float mx = halfmax(fmaxf(s0[r], s1[r]));
      float mnew = fmaxf(mI[r], mx);
      float a = __expf(mI[r] - mnew);
      mI[r] = mnew;
      float p0 = __expf(s0[r] - mnew);
      float p1 = __expf(s1[r] - mnew);
      lI[r] = lI[r] * a + halfsum(p0 + p1);
      plds[w][r + 8 * h][c]      = (_Float16)p0;  // C-layout -> LDS
      plds[w][r + 8 * h][16 + c] = (_Float16)p1;
      o0[r] *= a; o1[r] *= a; o2[r] *= a; o3[r] *= a;
    }
    asm volatile("" ::: "memory");  // keep LDS store->load order

    // ---- O += P @ V : P as A-frag from LDS, V from LDS V^T tile ----
    const v16h pf = ldA(&plds[w][c][0], h);
    const _Float16* vr = &vl[cur][c][0] + 16 * h;
    o0 = wmma(pf, ldB(vr), o0);
    o1 = wmma(pf, ldB(vr + 16 * 32), o1);
    o2 = wmma(pf, ldB(vr + 32 * 32), o2);
    o3 = wmma(pf, ldB(vr + 48 * 32), o3);

    cur ^= 1;
  }

  // ---- epilogue: O /= l, write f16 [B*QLEN, EMB] ----
#pragma unroll
  for (int r = 0; r < 8; r++) {
    float inv = 1.0f / lI[r];
    _Float16* orow = ao + (size_t)(qrow0 + r + 8 * h) * EMB + hcol;
    orow[c]      = (_Float16)(o0[r] * inv);
    orow[16 + c] = (_Float16)(o1[r] * inv);
    orow[32 + c] = (_Float16)(o2[r] * inv);
    orow[48 + c] = (_Float16)(o3[r] * inv);
  }
}

// ---------------------------------------------------------------------------
extern "C" void kernel_launch(void* const* d_in, const int* in_sizes, int n_in,
                              void* d_out, int out_size, void* d_ws, size_t ws_size,
                              hipStream_t stream) {
  const float* x   = (const float*)d_in[0];
  const float* ctx = (const float*)d_in[1];
  const float* Wq  = (const float*)d_in[2];
  const float* Wk  = (const float*)d_in[3];
  const float* Wv  = (const float*)d_in[4];
  const float* Wo  = (const float*)d_in[5];
  const float* bo  = (const float*)d_in[6];
  float* out = (float*)d_out;

  _Float16* ws  = (_Float16*)d_ws;
  _Float16* xb  = ws + OFF_XB;
  _Float16* cb  = ws + OFF_CB;
  _Float16* wqb = ws + OFF_WQ;
  _Float16* wkb = ws + OFF_WK;
  _Float16* wvb = ws + OFF_WV;
  _Float16* wob = ws + OFF_WO;
  _Float16* qbf = ws + OFF_QB;
  _Float16* kbf = ws + OFF_KB;
  _Float16* vtf = ws + OFF_VT;
  _Float16* aof = ws + OFF_AO;

  auto cvt = [&](const float* in, _Float16* o, size_t n) {
    int n4 = (int)(n / 4);
    cvt_f32_f16<<<(n4 + 255) / 256, 256, 0, stream>>>(in, o, n4);
  };
  cvt(x,   xb,  XB_N);
  cvt(ctx, cb,  CB_N);
  cvt(Wq,  wqb, W_N);
  cvt(Wk,  wkb, W_N);
  cvt(Wv,  wvb, W_N);
  cvt(Wo,  wob, W_N);

  const int Mq = BATCH * QLEN;    // 4096
  const int Mk = BATCH * KVLEN;   // 8192
  gemm_wmma<0><<<(Mq / 32) * (EMB / 256), 256, 0, stream>>>(xb, wqb, qbf, nullptr, nullptr, Mq, EMB);
  gemm_wmma<0><<<(Mk / 32) * (EMB / 256), 256, 0, stream>>>(cb, wkb, kbf, nullptr, nullptr, Mk, EMB);
  gemm_wmma<1><<<(Mk / 32) * (EMB / 256), 256, 0, stream>>>(cb, wvb, vtf, nullptr, nullptr, Mk, EMB);
  attn_wmma<<<BATCH * HEADS * (QLEN / 128), 256, 0, stream>>>(qbf, kbf, vtf, aof);
  gemm_wmma<2><<<(Mq / 32) * (EMB / 256), 256, 0, stream>>>(aof, wob, nullptr, out, bo, Mq, EMB);
}
